// Attn_5102421147813
// MI455X (gfx1250) — matrix operand order repaired and checked by reading, not compiled, not run on hardware
//
#include <hip/hip_runtime.h>
#include <math.h>

// ---------------------------------------------------------------- types ----
typedef __attribute__((ext_vector_type(16))) __bf16 v16bf;
typedef __attribute__((ext_vector_type(8)))  __bf16 bf16x8;
typedef __attribute__((ext_vector_type(8)))  float  v8f;
typedef __attribute__((ext_vector_type(4)))  unsigned int v4u;
typedef __attribute__((ext_vector_type(8)))  int v8i;
typedef __attribute__((ext_vector_type(4)))  int v4i;

#define BATCH 2
#define T     2048
#define CDIM  1024
#define NH    16
#define HD    64
#define BT    (BATCH * T)          // 4096 rows
#define ATT_SCALE 0.125f           // 1/sqrt(64)

// ------------------------------------------------------------- helpers ----
__device__ __forceinline__ v8f wmma_bf16(v16bf a, v16bf b, v8f c) {
  // D = A(16x32 bf16) * B(32x16 bf16) + C(16x16 f32)
  return __builtin_amdgcn_wmma_f32_16x16x32_bf16(
      /*neg_a=*/false, a, /*neg_b=*/false, b,
      /*c_mod=*/(short)0, c, /*reuse_a=*/false, /*reuse_b=*/false);
}

// A-fragment (16x32, bf16): lane = 16*half + m ; element e holds
// k = half*16 + e (callers pass c1 = c0 + 8). Per lane: two contiguous 8-elem runs.
__device__ __forceinline__ v16bf load_afrag(const __bf16* rowp, int c0, int c1) {
  bf16x8 l0 = *(const bf16x8*)(rowp + c0);
  bf16x8 l1 = *(const bf16x8*)(rowp + c1);
  v16bf r;
#pragma unroll
  for (int e = 0; e < 8; ++e) { r[e] = l0[e]; r[8 + e] = l1[e]; }
  return r;
}

// TDM: DMA one 128-row x 32-col bf16 tile (row stride 1024 elems) from a
// 4096x1024 bf16 tensor into LDS (row-major, 8 KB). ISA cdna5 §8 D# layout.
// This toolchain exposes the 6-arg builtin:
//   (uint32x4 g0, int32x8 g1, int32x4 g2, int32x4 g3, int32x8, i32 cpol)
__device__ __forceinline__ void tdm_load_tile(const __bf16* gsrc,
                                              unsigned lds_off) {
  unsigned long long ga = (unsigned long long)(uintptr_t)gsrc;
  v4u g0;
  g0[0] = 1u;                                   // count=1, user mode
  g0[1] = lds_off;                              // lds_addr (bytes)
  g0[2] = (unsigned)(ga & 0xFFFFFFFFu);         // global_addr[31:0]
  g0[3] = (unsigned)((ga >> 32) & 0x01FFFFFFu)  // global_addr[56:32]
          | 0x80000000u;                        // type = 2 ("image")
  v8i g1;
  g1[0] = 0x00010000;                           // data_size = 2 bytes
  g1[1] = (int)(1024u << 16);                   // tensor_dim0 = 1024 (bits 63:48)
  g1[2] = (int)(4096u << 16);                   // tensor_dim1 = 4096 (bits 95:80)
  g1[3] = (int)(32u << 16);                     // tile_dim0 = 32   (bits 127:112)
  g1[4] = 128;                                  // tile_dim1 = 128  (bits 143:128)
  g1[5] = 1024;                                 // tensor_dim0_stride (bits 191:160)
  g1[6] = 0;
  g1[7] = 0;
  v4i z4 = {0, 0, 0, 0};                        // 2-D tile: groups 2/3 unused
  v8i z8 = {0, 0, 0, 0, 0, 0, 0, 0};
  __builtin_amdgcn_tensor_load_to_lds(g0, g1, z4, z4, z8, 0);
}

// ---------------------------------------------------------- pack kernels ----
__global__ void cvt_f32_to_bf16(const float* __restrict__ x,
                                __bf16* __restrict__ y, int n) {
  int i = blockIdx.x * blockDim.x + threadIdx.x;
  if (i < n) y[i] = (__bf16)x[i];
}

// Pack nn.Linear weight W [out][in] (f32) directly into WMMA B-fragment
// lane order for Y = X @ W^T:  B(k=in, n=out).
// Packed index: (((kg*64 + nt)*32 + lane)*16 + e)
//   with n = nt*16 + lane%16 ,  k = kg*32 + (lane/16)*16 + e.
// GEMM then loads B as one contiguous 32B/lane read.
__global__ void pack_w_bfrag(const float* __restrict__ w,
                             __bf16* __restrict__ wp) {
  int i = blockIdx.x * blockDim.x + threadIdx.x;   // 0 .. 1024*1024-1
  int e    = i & 15;
  int lane = (i >> 4) & 31;
  int nt   = (i >> 9) & 63;
  int kg   = i >> 15;                              // 0..31
  int n = nt * 16 + (lane & 15);
  int k = kg * 32 + (lane >> 4) * 16 + e;
  wp[i] = (__bf16)w[(size_t)n * CDIM + k];
}

// ------------------------------------------------------------ WMMA GEMM ----
// Y[4096,1024] = X[4096,1024](bf16) @ W^T, W packed as B-fragments (wp).
// Block tile 128(M) x 128(N), K-step 32, 256 threads = 8 waves.
// Wave w owns m-tile w and all 8 n-tiles: 8 accumulators, 8 WMMA / K-step.
// X tiles DMA'd into double-buffered LDS by the Tensor Data Mover (wave 0
// issues TENSOR_LOAD_TO_LDS, synced with s_wait_tensorcnt + barrier); B read
// straight from L2-resident packed weights (contiguous 32B per lane).
// mode 0: f32 natural [row, col]   (final output)
// mode 1: bf16 [b, h, t, d]        (Q, K)
// mode 2: bf16 [b, h, d, t]        (V transposed)
#define GM 128
#define GN 128
#define GK 32
#define NKG (CDIM / GK)   // 32 K-steps

__global__ __launch_bounds__(256) void gemm_bf16(
    const __bf16* __restrict__ X, const __bf16* __restrict__ Wp,
    void* __restrict__ outp, int mode) {
  __shared__ __bf16 Xs[2][GM * GK];   // 2 x 8 KB, TDM destination

  const int tid    = threadIdx.x;
  const int lane   = tid & 31;
  const int wm     = tid >> 5;        // wave id = m-tile 0..7
  const int nprime = lane & 15;
  const int half   = lane >> 4;
  const int mBase  = blockIdx.x * GM;
  const int ntg0   = blockIdx.y * 8;  // global n-tile base (8 tiles/block)
  const bool issuer = (wm == 0);      // wave 0 drives the TDM

  const __bf16* xbase = X + (size_t)mBase * CDIM;   // tile kg starts at +kg*GK
  const unsigned ldsA = (unsigned)(uintptr_t)&Xs[0][0];
  const unsigned ldsB = (unsigned)(uintptr_t)&Xs[1][0];

  v8f acc[8];
#pragma unroll
  for (int i = 0; i < 8; ++i) acc[i] = v8f{};

  // prime buffer 0 via TDM
  if (issuer) {
    tdm_load_tile(xbase, ldsA);
    __builtin_amdgcn_s_wait_tensorcnt(0);
  }
  __syncthreads();

  for (int kg = 0; kg < NKG; ++kg) {
    const int cur = kg & 1;
    // DMA next X tile into the other buffer (overlaps WMMAs below)
    if (issuer && (kg + 1 < NKG))
      tdm_load_tile(xbase + (size_t)(kg + 1) * GK, cur ? ldsA : ldsB);

    // A fragment from LDS: rows 16*wm .. +15
    v16bf a = load_afrag(&Xs[cur][(wm * 16 + nprime) * GK],
                         half * 16, half * 16 + 8);

    // 8 B fragments straight from packed weights (L2), 8 back-to-back WMMAs
    const __bf16* wpb = Wp + (((size_t)kg * 64 + ntg0) * 32 + lane) * 16;
#pragma unroll
    for (int nt = 0; nt < 8; ++nt) {
      v16bf b = *(const v16bf*)(wpb + (size_t)nt * 512);
      acc[nt] = wmma_bf16(a, b, acc[nt]);
    }

    if (issuer && (kg + 1 < NKG))
      __builtin_amdgcn_s_wait_tensorcnt(0);   // next tile landed in LDS
    __syncthreads();                          // + everyone done reading `cur`
  }

  // epilogue: C layout -> row = 16*wm + r + 8*half, col = lane%16
  const int gm0 = mBase + wm * 16;
#pragma unroll
  for (int nt = 0; nt < 8; ++nt) {
    const int col = (ntg0 + nt) * 16 + nprime;
#pragma unroll
    for (int r = 0; r < 8; ++r) {
      const int gr = gm0 + r + 8 * half;
      const float v = acc[nt][r];
      if (mode == 0) {
        ((float*)outp)[(size_t)gr * CDIM + col] = v;
      } else {
        const int b = gr >> 11, t = gr & (T - 1);
        const int h = col >> 6, d = col & (HD - 1);
        if (mode == 1)
          ((__bf16*)outp)[((size_t)(b * NH + h) * T + t) * HD + d] = (__bf16)v;
        else
          ((__bf16*)outp)[((size_t)(b * NH + h) * HD + d) * T + t] = (__bf16)v;
      }
    }
  }
}

// ------------------------------------------------------ flash attention ----
// One wave per 16-query tile, per (b,h). Computes in transposed orientation:
//   S^T = K * Q^T     (A = K rows, B = Q rows)
//   O^T = V^T * P^T   (A = V^T rows, B = exp(S^T) via xor-16 lane shuffle)
// Q: [b,h,t,d] bf16    K: [b,h,t,d] bf16    Vt: [b,h,d,t] bf16
// O: [b*T, C] bf16 (natural layout, feeds the final GEMM)
__global__ __launch_bounds__(256) void attn_fwd(
    const __bf16* __restrict__ Q, const __bf16* __restrict__ K,
    const __bf16* __restrict__ Vt, __bf16* __restrict__ O) {
  const int tid  = threadIdx.x;
  const int lane = tid & 31;
  const int wid  = blockIdx.x * (blockDim.x >> 5) + (tid >> 5);  // 0..4095
  const int qt   = wid & 127;   // 16-row query tile within (b,h)
  const int bh   = wid >> 7;    // 0..31
  const int qb   = qt * 16;

  const __bf16* Qp = Q  + (size_t)bh * T * HD;
  const __bf16* Kp = K  + (size_t)bh * T * HD;
  const __bf16* Vp = Vt + (size_t)bh * HD * T;

  const int  nprime = lane & 15;
  const int  half   = lane >> 4;
  const bool hi     = half != 0;

  // Persistent Q B-fragments: lane col q = qb + nprime, d = d0 + half*16 + e
  const __bf16* qrow = Qp + (size_t)(qb + nprime) * HD + half * 16;
  const v16bf qf0 = *(const v16bf*)qrow;          // d 0..31
  const v16bf qf1 = *(const v16bf*)(qrow + 32);   // d 32..63

  float m_run = -INFINITY;
  float l_run = 0.0f;
  v8f oc[4] = {v8f{}, v8f{}, v8f{}, v8f{}};       // O^T tiles, d-tiles 0..3

  const int ngroups = (qb + 16 + 31) >> 5;        // ceil((qb+16)/32) key groups
  for (int g = 0; g < ngroups; ++g) {
    const int kb = g * 32;

    // --- S^T: two 16x16 tiles (keys kb..kb+15 and kb+16..kb+31) ---
    v8f s0 = {}, s1 = {};
    {
      const __bf16* r0 = Kp + (size_t)(kb + nprime) * HD;
      const __bf16* r1 = Kp + (size_t)(kb + 16 + nprime) * HD;
      v16bf ka;
      ka = load_afrag(r0, half * 16, half * 16 + 8);       s0 = wmma_bf16(ka, qf0, s0);
      ka = load_afrag(r0, 32 + half * 16, 40 + half * 16);  s0 = wmma_bf16(ka, qf1, s0);
      ka = load_afrag(r1, half * 16, half * 16 + 8);       s1 = wmma_bf16(ka, qf0, s1);
      ka = load_afrag(r1, 32 + half * 16, 40 + half * 16);  s1 = wmma_bf16(ka, qf1, s1);
    }

    // --- scale + causal mask + column(=query) max ---
    const int qg = qb + nprime;
    float cmax = -INFINITY;
#pragma unroll
    for (int r = 0; r < 8; ++r) {
      const int k0 = kb + r + 8 * half;
      const int k1 = k0 + 16;
      float v0 = s0[r] * ATT_SCALE; if (k0 > qg) v0 = -INFINITY;
      float v1 = s1[r] * ATT_SCALE; if (k1 > qg) v1 = -INFINITY;
      s0[r] = v0; s1[r] = v1;
      cmax = fmaxf(cmax, fmaxf(v0, v1));
    }
    cmax = fmaxf(cmax, __shfl_xor(cmax, 16, 32));

    const float m_new = fmaxf(m_run, cmax);
    const float alpha = __expf(m_run - m_new);
    m_run = m_new;

    // --- P = exp(S - m), column sums ---
    float psum = 0.0f;
#pragma unroll
    for (int r = 0; r < 8; ++r) {
      const float p0 = __expf(s0[r] - m_new);
      const float p1 = __expf(s1[r] - m_new);
      s0[r] = p0; s1[r] = p1;
      psum += p0 + p1;
    }
    psum += __shfl_xor(psum, 16, 32);
    l_run = l_run * alpha + psum;

    // --- P^T as B-fragment: C-layout -> B-layout via xor-16 shuffle ---
    v16bf pb;
#pragma unroll
    for (int r = 0; r < 8; ++r) {
      const float x0 = s0[r], x1 = s1[r];
      const float y0 = __shfl_xor(x0, 16, 32);
      const float y1 = __shfl_xor(x1, 16, 32);
      pb[r]     = (__bf16)(hi ? y1 : x0);   // k = half*16 + r
      pb[8 + r] = (__bf16)(hi ? x1 : y0);   // k = half*16 + 8 + r
    }

    // --- rescale running O^T, then O^T += V^T * P^T ---
#pragma unroll
    for (int dt = 0; dt < 4; ++dt) {
#pragma unroll
      for (int r = 0; r < 8; ++r) oc[dt][r] *= alpha;
    }
#pragma unroll
    for (int dt = 0; dt < 4; ++dt) {
      const __bf16* vrow = Vp + (size_t)(dt * 16 + nprime) * T + kb;
      v16bf va = load_afrag(vrow, half * 16, half * 16 + 8);
      oc[dt] = wmma_bf16(va, pb, oc[dt]);
    }
  }

  // --- normalize and store O (natural [b*T, C] bf16) ---
  const float inv = 1.0f / l_run;
  const int b = bh >> 4, h = bh & 15;
  const size_t rowoff = (size_t)(b * T + qb + nprime) * CDIM + h * HD;
#pragma unroll
  for (int dt = 0; dt < 4; ++dt) {
#pragma unroll
    for (int r = 0; r < 8; ++r)
      O[rowoff + dt * 16 + r + 8 * half] = (__bf16)(oc[dt][r] * inv);
  }
}

// --------------------------------------------------------------- launch ----
extern "C" void kernel_launch(void* const* d_in, const int* in_sizes, int n_in,
                              void* d_out, int out_size, void* d_ws,
                              size_t ws_size, hipStream_t stream) {
  (void)in_sizes; (void)n_in; (void)out_size; (void)ws_size;
  const float* x  = (const float*)d_in[0];
  const float* wq = (const float*)d_in[1];
  const float* wk = (const float*)d_in[2];
  const float* wv = (const float*)d_in[3];
  const float* wo = (const float*)d_in[4];

  char* ws = (char*)d_ws;
  const size_t SZ_XB = (size_t)BT * CDIM * sizeof(__bf16);           // 8 MB
  const size_t SZ_W  = (size_t)CDIM * CDIM * sizeof(__bf16);         // 2 MB
  const size_t SZ_H  = (size_t)BATCH * NH * T * HD * sizeof(__bf16); // 8 MB
  size_t off = 0;
  __bf16* xb  = (__bf16*)(ws + off); off += SZ_XB;
  __bf16* wqP = (__bf16*)(ws + off); off += SZ_W;
  __bf16* wkP = (__bf16*)(ws + off); off += SZ_W;
  __bf16* wvP = (__bf16*)(ws + off); off += SZ_W;
  __bf16* woP = (__bf16*)(ws + off); off += SZ_W;
  __bf16* Qb  = (__bf16*)(ws + off); off += SZ_H;
  __bf16* Kb  = (__bf16*)(ws + off); off += SZ_H;
  __bf16* Vtb = (__bf16*)(ws + off); off += SZ_H;
  __bf16* Ob  = (__bf16*)(ws + off); off += SZ_XB;

  // 1) convert x; pack weights into WMMA B-fragment order (bf16)
  cvt_f32_to_bf16<<<(BT * CDIM) / 256, 256, 0, stream>>>(x, xb, BT * CDIM);
  pack_w_bfrag<<<(CDIM * CDIM) / 256, 256, 0, stream>>>(wq, wqP);
  pack_w_bfrag<<<(CDIM * CDIM) / 256, 256, 0, stream>>>(wk, wkP);
  pack_w_bfrag<<<(CDIM * CDIM) / 256, 256, 0, stream>>>(wv, wvP);
  pack_w_bfrag<<<(CDIM * CDIM) / 256, 256, 0, stream>>>(wo, woP);

  // 2) projections (WMMA GEMMs), 128x128 block tiles, TDM-staged X
  dim3 ggrid(BT / GM, CDIM / GN);   // 32 x 8
  gemm_bf16<<<ggrid, 256, 0, stream>>>(xb, wqP, Qb, 1);   // Q [b,h,t,d]
  gemm_bf16<<<ggrid, 256, 0, stream>>>(xb, wkP, Kb, 1);   // K [b,h,t,d]
  gemm_bf16<<<ggrid, 256, 0, stream>>>(xb, wvP, Vtb, 2);  // V^T [b,h,d,t]

  // 3) causal flash attention: 4096 waves, 8 waves/block
  attn_fwd<<<(BATCH * NH * (T / 16)) / 8, 256, 0, stream>>>(Qb, Kb, Vtb, Ob);

  // 4) output projection -> f32 d_out
  gemm_bf16<<<ggrid, 256, 0, stream>>>(Ob, woP, d_out, 0);
}
